// BottleneckResAtnMHSA_21449066676839
// MI455X (gfx1250) — matrix operand order repaired and checked by reading, not compiled, not run on hardware
//
#include <hip/hip_runtime.h>
#include <cstdint>

typedef __attribute__((ext_vector_type(16))) __bf16 v16bf;
typedef __attribute__((ext_vector_type(8)))  float  v8f;
typedef unsigned short u16;
typedef unsigned int   u32;
typedef unsigned long long u64;

#define DEV static __device__ __forceinline__

constexpr int BATCH = 32;
constexpr int CIN   = 256;
constexpr int C     = 128;   // bottleneck channels
constexpr int NSP   = 1024;  // H*W

struct __align__(16) U4 { u32 x, y, z, w; };
struct __align__(8)  F2 { float x, y; };
struct __align__(16) F4 { float x, y, z, w; };

// float -> bf16 bits, round-to-nearest-even
DEV u16 f2bf(float f){
  u32 u = __builtin_bit_cast(u32, f);
  return (u16)((u + 0x7FFFu + ((u >> 16) & 1u)) >> 16);
}
DEV u32 pk2(float a, float b){ return (u32)f2bf(a) | ((u32)f2bf(b) << 16); }
DEV float silu(float v){ return v / (1.0f + __expf(-v)); }

// CDNA5 async global->LDS copy, 16 bytes per lane. Tracked by ASYNCcnt.
// VDST vgpr = LDS byte offset (generic LDS addr low 32 bits), VADDR = 64-bit global.
DEV void async_cp16(u32 lds_off, const void* gsrc){
  u64 ga = (u64)(uintptr_t)gsrc;
  asm volatile("global_load_async_to_lds_b128 %0, %1, off"
               :: "v"(lds_off), "v"(ga) : "memory");
}
DEV void async_wait0(){
  asm volatile("s_wait_asynccnt 0x0" ::: "memory");
}

// Build a 16-element bf16 fragment from two 16-byte chunks.
DEV v16bf frag2(const u16* p0, const u16* p1){
  v16bf v;
  u32* d = reinterpret_cast<u32*>(&v);
  const u32* a = reinterpret_cast<const u32*>(p0);
  const u32* b = reinterpret_cast<const u32*>(p1);
  d[0]=a[0]; d[1]=a[1]; d[2]=a[2]; d[3]=a[3];
  d[4]=b[0]; d[5]=b[1]; d[6]=b[2]; d[7]=b[3];
  return v;
}
// A fragment (16x32 bf16), storage row-major [m][k], row stride ldk elements.
// VGPR p: K = 16*(lane>=16) + 2p+e -> one 16-element run.
DEV v16bf load_a(const u16* base, int ldk, int lane){
  const u16* p = base + (lane & 15) * ldk + ((lane >> 4) << 4);
  return frag2(p, p + 8);
}
// B fragment (32x16 bf16), storage [n][k] (k contiguous), row stride ldk.
// VGPR p: K = 16*(lane>=16) + 2p+e -> one 16-element run.
DEV v16bf load_b(const u16* base, int ldk, int lane){
  const u16* p = base + (lane & 15) * ldk + ((lane >> 4) << 4);
  return frag2(p, p + 8);
}
DEV v8f wmma_bf16(v16bf a, v16bf b, v8f c){
  return __builtin_amdgcn_wmma_f32_16x16x32_bf16(false, a, false, b, (short)0, c, false, false);
}
DEV v8f vzero(){ v8f z = {0.f,0.f,0.f,0.f,0.f,0.f,0.f,0.f}; return z; }

// ---------------------------------------------------------------------------
// Kernel 0: BN scale/shift precompute + positional table pos_t[i][c] (bf16)
// ---------------------------------------------------------------------------
__global__ __launch_bounds__(256)
void prep_kernel(const float* __restrict__ g1, const float* __restrict__ b1,
                 const float* __restrict__ m1, const float* __restrict__ v1,
                 const float* __restrict__ g2, const float* __restrict__ b2,
                 const float* __restrict__ m2, const float* __restrict__ v2,
                 const float* __restrict__ relh, const float* __restrict__ relw,
                 float* __restrict__ bn1s, float* __restrict__ bn1t,
                 float* __restrict__ bn2s, float* __restrict__ bn2t,
                 u16* __restrict__ pos_t){
  int t = blockIdx.x * 256 + threadIdx.x;
  if (t < C){
    float s = g1[t] * rsqrtf(v1[t] + 1e-5f);
    bn1s[t] = s; bn1t[t] = b1[t] - m1[t] * s;
  } else if (t < C + CIN){
    int c = t - C;
    float s = g2[c] * rsqrtf(v2[c] + 1e-5f);
    bn2s[c] = s; bn2t[c] = b2[c] - m2[c] * s;
  }
  int c = t & (C - 1);
  int i = t >> 7;                      // 0..1023
  pos_t[(size_t)i * C + c] = f2bf(relh[c * 32 + (i >> 5)] + relw[c * 32 + (i & 31)]);
}

// ---------------------------------------------------------------------------
// Kernel 1: cv1  x1 = SiLU(BN(w1 @ x)), output transposed bf16 x1_t[b][n][c]
// Block: (n-slab of 64, batch). 8 waves, wave w owns M strip [16w,16w+16).
// (x slab needs transpose+fp32->bf16 conversion, so it stays on the VALU path.)
// ---------------------------------------------------------------------------
__global__ __launch_bounds__(256)
void cv1_kernel(const float* __restrict__ x, const float* __restrict__ w1,
                const float* __restrict__ bn1s, const float* __restrict__ bn1t,
                u16* __restrict__ x1_t){
  __shared__ __align__(16) u32 wsh[C * CIN / 2];     // 64 KB: w1 as bf16 [c][cin]
  __shared__ __align__(16) u16 xts[64 * CIN];        // 32 KB: x slab transposed [n][cin]
  int tid = threadIdx.x;
  int b = blockIdx.y, n0 = blockIdx.x * 64;

  for (int idx = tid; idx < C * CIN / 2; idx += 256){
    F2 f = reinterpret_cast<const F2*>(w1)[idx];
    wsh[idx] = pk2(f.x, f.y);
  }
  {
    int n = tid & 63, c0 = tid >> 6;
    const float* xb = x + (((size_t)b * CIN) << 10) + n0 + n;
    for (int cin = c0; cin < CIN; cin += 4)
      xts[n * CIN + cin] = f2bf(xb[(size_t)cin << 10]);
  }
  __syncthreads();

  int lane = tid & 31, wave = tid >> 5;
  int m0 = wave * 16, half = lane >> 4, nl = lane & 15;
  const u16* wrow = reinterpret_cast<const u16*>(wsh) + m0 * CIN;

  v8f acc[4] = {vzero(), vzero(), vzero(), vzero()};
  for (int kc = 0; kc < CIN / 32; ++kc){
    v16bf a = load_a(wrow + kc * 32, CIN, lane);
#pragma unroll
    for (int t = 0; t < 4; ++t){
      v16bf bf = load_b(xts + (t * 16) * CIN + kc * 32, CIN, lane);
      acc[t] = wmma_bf16(a, bf, acc[t]);
    }
  }

  float sc[8], sh[8];
#pragma unroll
  for (int r = 0; r < 8; ++r){
    int c = m0 + half * 8 + r;
    sc[r] = bn1s[c]; sh[r] = bn1t[c];
  }
#pragma unroll
  for (int t = 0; t < 4; ++t){
    int n = n0 + t * 16 + nl;
    float f[8];
#pragma unroll
    for (int r = 0; r < 8; ++r) f[r] = silu(acc[t][r] * sc[r] + sh[r]);
    U4 o = { pk2(f[0], f[1]), pk2(f[2], f[3]), pk2(f[4], f[5]), pk2(f[6], f[7]) };
    *reinterpret_cast<U4*>(x1_t + ((size_t)((b << 10) + n)) * C + m0 + half * 8) = o;
  }
}

// ---------------------------------------------------------------------------
// Kernel 2: q/k/v GEMMs sharing one x1 slab (async-DMA'd into LDS).
// q,k stored transposed [b][n][c]; v stored natural [b][c][n] (A-side of PV).
// ---------------------------------------------------------------------------
__global__ __launch_bounds__(256)
void qkv_kernel(const u16* __restrict__ x1_t,
                const float* __restrict__ wq, const float* __restrict__ bq,
                const float* __restrict__ wk, const float* __restrict__ bk,
                const float* __restrict__ wv, const float* __restrict__ bv,
                u16* __restrict__ q_t, u16* __restrict__ k_t, u16* __restrict__ v_n){
  __shared__ __align__(16) u32 wsh[3][C * C / 2];    // 96 KB
  __shared__ __align__(16) u32 xsh[64 * C / 2];      // 16 KB: x1 slab [n][c] bf16
  int tid = threadIdx.x;
  int b = blockIdx.y, n0 = blockIdx.x * 64;

  // Async DMA the bf16 x1 slab straight into LDS (no VGPR round trip),
  // overlapping with the fp32->bf16 weight conversion below.
  {
    const char* gsrc = reinterpret_cast<const char*>(x1_t + ((size_t)((b << 10) + n0)) * C);
    for (int pkt = tid; pkt < 64 * C * 2 / 16; pkt += 256)       // 1024 x 16B
      async_cp16((u32)(uintptr_t)xsh + pkt * 16, gsrc + pkt * 16);
  }
  const float* wl[3] = {wq, wk, wv};
  for (int s = 0; s < 3; ++s)
    for (int idx = tid; idx < C * C / 2; idx += 256){
      F2 f = reinterpret_cast<const F2*>(wl[s])[idx];
      wsh[s][idx] = pk2(f.x, f.y);
    }
  async_wait0();
  __syncthreads();

  int lane = tid & 31, wave = tid >> 5;
  int m0 = wave * 16, half = lane >> 4, nl = lane & 15;
  const float* bl[3] = {bq, bk, bv};
  u16* outl[2] = {q_t, k_t};

  for (int s = 0; s < 3; ++s){
    v8f acc[4] = {vzero(), vzero(), vzero(), vzero()};
    const u16* wrow = reinterpret_cast<const u16*>(wsh[s]) + m0 * C;
#pragma unroll
    for (int kc = 0; kc < C / 32; ++kc){
      v16bf a = load_a(wrow + kc * 32, C, lane);
#pragma unroll
      for (int t = 0; t < 4; ++t){
        v16bf bf = load_b(reinterpret_cast<const u16*>(xsh) + (t * 16) * C + kc * 32, C, lane);
        acc[t] = wmma_bf16(a, bf, acc[t]);
      }
    }
    float bias[8];
#pragma unroll
    for (int r = 0; r < 8; ++r) bias[r] = bl[s][m0 + half * 8 + r];

    if (s < 2){
      u16* dst = outl[s];
#pragma unroll
      for (int t = 0; t < 4; ++t){
        int n = n0 + t * 16 + nl;
        float f[8];
#pragma unroll
        for (int r = 0; r < 8; ++r) f[r] = acc[t][r] + bias[r];
        U4 o = { pk2(f[0], f[1]), pk2(f[2], f[3]), pk2(f[4], f[5]), pk2(f[6], f[7]) };
        *reinterpret_cast<U4*>(dst + ((size_t)((b << 10) + n)) * C + m0 + half * 8) = o;
      }
    } else {
#pragma unroll
      for (int t = 0; t < 4; ++t){
        int n = n0 + t * 16 + nl;
#pragma unroll
        for (int r = 0; r < 8; ++r){
          int c = m0 + half * 8 + r;
          v_n[(((size_t)b * C + c) << 10) + n] = f2bf(acc[t][r] + bias[r]);
        }
      }
    }
  }
}

// ---------------------------------------------------------------------------
// Kernel 3: fused attention for one (batch, 16-row i-block).
// Phase 1: S[16][1024] = [q_t|pos_t] x [k_t|q_t]^T  (K=256)  -> LDS fp32
// Phase 2: row softmax (wave32 shuffle reductions)          -> LDS bf16 P
// Phase 3: out_t[i][c] = sum_j v[c][j] P[i][j]  (K=1024)
// ---------------------------------------------------------------------------
__global__ __launch_bounds__(256)
void attn_kernel(const u16* __restrict__ q_t, const u16* __restrict__ k_t,
                 const u16* __restrict__ v_n, const u16* __restrict__ pos_t,
                 u16* __restrict__ out_t){
  __shared__ __align__(16) u16  ash[16 * 256];       //  8 KB: A rows = q_t[i] || pos_t[i]
  __shared__ __align__(16) float ssh[16 * NSP];      // 64 KB: logits
  __shared__ __align__(16) u16  psh[16 * NSP];       // 32 KB: probabilities (bf16)
  int tid = threadIdx.x;
  int b = blockIdx.y, i0 = blockIdx.x * 16;

  // Async DMA the A block: per i-row, 256B of q_t then 256B of pos_t.
  for (int pkt = tid; pkt < 512; pkt += 256){
    int row  = pkt >> 5;          // 0..15
    int part = (pkt >> 4) & 1;    // 0 = q_t, 1 = pos_t
    int seg  = pkt & 15;          // 16B segment within 256B
    int i = i0 + row;
    const u16* src = part ? (pos_t + (size_t)i * C)
                          : (q_t + ((size_t)((b << 10) + i)) * C);
    async_cp16((u32)(uintptr_t)ash + row * 512 + part * 256 + seg * 16,
               reinterpret_cast<const char*>(src) + seg * 16);
  }
  async_wait0();
  __syncthreads();

  int lane = tid & 31, wave = tid >> 5, half = lane >> 4, nl = lane & 15;

  v16bf afr[8];
#pragma unroll
  for (int kc = 0; kc < 8; ++kc) afr[kc] = load_a(ash + kc * 32, 256, lane);

  for (int jt = wave; jt < 64; jt += 8){
    int j0 = jt * 16;
    v8f acc = vzero();
    const u16* kb = k_t + ((size_t)((b << 10) + j0)) * C;
    const u16* qb = q_t + ((size_t)((b << 10) + j0)) * C;
#pragma unroll
    for (int kc = 0; kc < 4; ++kc) acc = wmma_bf16(afr[kc],     load_b(kb + kc * 32, C, lane), acc);
#pragma unroll
    for (int kc = 0; kc < 4; ++kc) acc = wmma_bf16(afr[4 + kc], load_b(qb + kc * 32, C, lane), acc);
#pragma unroll
    for (int r = 0; r < 8; ++r) ssh[(half * 8 + r) * NSP + j0 + nl] = acc[r];
  }
  __syncthreads();

  // Softmax: each wave handles rows {wave, wave+8}; lane owns 32 contiguous cols.
  for (int rr = 0; rr < 2; ++rr){
    int row = wave + rr * 8;
    const F4* srow = reinterpret_cast<const F4*>(ssh + row * NSP + lane * 32);
    float e[32];
    float mx = -3.0e38f;
#pragma unroll
    for (int q4 = 0; q4 < 8; ++q4){
      F4 f = srow[q4];
      e[q4*4+0] = f.x; e[q4*4+1] = f.y; e[q4*4+2] = f.z; e[q4*4+3] = f.w;
      mx = fmaxf(mx, fmaxf(fmaxf(f.x, f.y), fmaxf(f.z, f.w)));
    }
#pragma unroll
    for (int o = 16; o > 0; o >>= 1) mx = fmaxf(mx, __shfl_xor(mx, o, 32));
    float sum = 0.f;
#pragma unroll
    for (int q2 = 0; q2 < 32; ++q2){ e[q2] = __expf(e[q2] - mx); sum += e[q2]; }
#pragma unroll
    for (int o = 16; o > 0; o >>= 1) sum += __shfl_xor(sum, o, 32);
    float inv = 1.f / sum;
    u32* prow = reinterpret_cast<u32*>(psh + row * NSP) + lane * 16;
#pragma unroll
    for (int kp = 0; kp < 16; ++kp)
      prow[kp] = pk2(e[2*kp] * inv, e[2*kp+1] * inv);
  }
  __syncthreads();

  // PV: D(m=c strip of 16, n=i block of 16), K = 1024.
  int m0 = wave * 16;
  const u16* vb = v_n + (((size_t)b * C + m0) << 10);
  v8f acc = vzero();
  for (int kc = 0; kc < 32; ++kc)
    acc = wmma_bf16(load_a(vb + kc * 32, NSP, lane),
                    load_b(psh + kc * 32, NSP, lane), acc);
  float f[8];
#pragma unroll
  for (int r = 0; r < 8; ++r) f[r] = acc[r];
  U4 o = { pk2(f[0], f[1]), pk2(f[2], f[3]), pk2(f[4], f[5]), pk2(f[6], f[7]) };
  *reinterpret_cast<U4*>(out_t + ((size_t)((b << 10) + i0 + nl)) * C + m0 + half * 8) = o;
}

// ---------------------------------------------------------------------------
// Kernel 4: cv2 + BN + SiLU + residual -> fp32 d_out [b][o][n]
// ---------------------------------------------------------------------------
__global__ __launch_bounds__(256)
void cv2_kernel(const u16* __restrict__ out_t, const float* __restrict__ w2,
                const float* __restrict__ bn2s, const float* __restrict__ bn2t,
                const float* __restrict__ x, float* __restrict__ y){
  __shared__ __align__(16) u32 wsh[CIN * C / 2];     // 64 KB: w2 bf16 [o][c]
  __shared__ __align__(16) u32 osh[64 * C / 2];      // 16 KB: out slab [n][c]
  int tid = threadIdx.x;
  int b = blockIdx.y, n0 = blockIdx.x * 64;

  // Async DMA the bf16 out_t slab into LDS, overlapped with w2 conversion.
  {
    const char* gsrc = reinterpret_cast<const char*>(out_t + ((size_t)((b << 10) + n0)) * C);
    for (int pkt = tid; pkt < 64 * C * 2 / 16; pkt += 256)       // 1024 x 16B
      async_cp16((u32)(uintptr_t)osh + pkt * 16, gsrc + pkt * 16);
  }
  for (int idx = tid; idx < CIN * C / 2; idx += 256){
    F2 f = reinterpret_cast<const F2*>(w2)[idx];
    wsh[idx] = pk2(f.x, f.y);
  }
  async_wait0();
  __syncthreads();

  int lane = tid & 31, wave = tid >> 5, half = lane >> 4, nl = lane & 15;
  for (int s = 0; s < 2; ++s){
    int m0 = wave * 16 + s * 128;
    const u16* wrow = reinterpret_cast<const u16*>(wsh) + m0 * C;
    v8f acc[4] = {vzero(), vzero(), vzero(), vzero()};
#pragma unroll
    for (int kc = 0; kc < C / 32; ++kc){
      v16bf a = load_a(wrow + kc * 32, C, lane);
#pragma unroll
      for (int t = 0; t < 4; ++t){
        v16bf bf = load_b(reinterpret_cast<const u16*>(osh) + (t * 16) * C + kc * 32, C, lane);
        acc[t] = wmma_bf16(a, bf, acc[t]);
      }
    }
    float sc[8], sh[8];
#pragma unroll
    for (int r = 0; r < 8; ++r){
      int oc = m0 + half * 8 + r;
      sc[r] = bn2s[oc]; sh[r] = bn2t[oc];
    }
#pragma unroll
    for (int t = 0; t < 4; ++t){
      int n = n0 + t * 16 + nl;
#pragma unroll
      for (int r = 0; r < 8; ++r){
        int oc = m0 + half * 8 + r;
        size_t idx = (((size_t)b * CIN + oc) << 10) + n;
        y[idx] = x[idx] + silu(acc[t][r] * sc[r] + sh[r]);
      }
    }
  }
}

// ---------------------------------------------------------------------------
extern "C" void kernel_launch(void* const* d_in, const int* in_sizes, int n_in,
                              void* d_out, int out_size, void* d_ws, size_t ws_size,
                              hipStream_t stream){
  (void)in_sizes; (void)n_in; (void)out_size; (void)ws_size;
  const float* x    = (const float*)d_in[0];
  const float* w1   = (const float*)d_in[1];
  const float* g1   = (const float*)d_in[2];
  const float* b1   = (const float*)d_in[3];
  const float* m1   = (const float*)d_in[4];
  const float* v1   = (const float*)d_in[5];
  const float* wq   = (const float*)d_in[6];
  const float* bq   = (const float*)d_in[7];
  const float* wk   = (const float*)d_in[8];
  const float* bk   = (const float*)d_in[9];
  const float* wv   = (const float*)d_in[10];
  const float* bv   = (const float*)d_in[11];
  const float* relh = (const float*)d_in[12];
  const float* relw = (const float*)d_in[13];
  const float* w2   = (const float*)d_in[14];
  const float* g2   = (const float*)d_in[15];
  const float* b2   = (const float*)d_in[16];
  const float* m2   = (const float*)d_in[17];
  const float* v2   = (const float*)d_in[18];

  // Workspace layout (all written before read each call; ~42.2 MB total)
  float* bn1s = (float*)d_ws;             // 128
  float* bn1t = bn1s + 128;               // 128
  float* bn2s = bn1t + 128;               // 256
  float* bn2t = bn2s + 256;               // 256
  u16* pos_t = (u16*)(bn2t + 256);                     // 1024*128
  u16* x1_t  = pos_t + (size_t)NSP * C;                // 32*1024*128
  u16* q_t   = x1_t + (size_t)BATCH * NSP * C;
  u16* k_t   = q_t  + (size_t)BATCH * NSP * C;
  u16* v_n   = k_t  + (size_t)BATCH * NSP * C;
  u16* out_t = v_n  + (size_t)BATCH * NSP * C;

  prep_kernel<<<512, 256, 0, stream>>>(g1, b1, m1, v1, g2, b2, m2, v2,
                                       relh, relw, bn1s, bn1t, bn2s, bn2t, pos_t);
  cv1_kernel<<<dim3(16, BATCH), 256, 0, stream>>>(x, w1, bn1s, bn1t, x1_t);
  qkv_kernel<<<dim3(16, BATCH), 256, 0, stream>>>(x1_t, wq, bq, wk, bk, wv, bv,
                                                  q_t, k_t, v_n);
  attn_kernel<<<dim3(64, BATCH), 256, 0, stream>>>(q_t, k_t, v_n, pos_t, out_t);
  cv2_kernel<<<dim3(16, BATCH), 256, 0, stream>>>(out_t, w2, bn2s, bn2t, x,
                                                  (float*)d_out);
}
